// SE_72688026517857
// MI455X (gfx1250) — matrix operand order repaired and checked by reading, not compiled, not run on hardware
//
#include <hip/hip_runtime.h>
#include <hip/hip_bf16.h>

#define B_   4
#define C_   128
#define H_   48
#define W_   48
#define N_   2304
#define NH_  4
#define DH_  32
#define HID_ 256
#define BN_  (B_ * N_)
#define BNC_ (B_ * N_ * C_)

typedef __attribute__((ext_vector_type(16))) __bf16 bf16x16;
typedef __attribute__((ext_vector_type(8)))  float  f32x8;

union FragBF {
    bf16x16 v;
    uint4   u[2];
    __bf16  e[16];
};

// --- CDNA5 async global->LDS copy (ASYNCcnt tracked), CDNA5 ISA ---
__device__ __forceinline__ void async_copy_b128(const void* g, void* l) {
#if defined(__HIP_DEVICE_COMPILE__)
    unsigned lds = (unsigned)(unsigned long long)l;   // low 32 bits of generic = LDS offset
    asm volatile("global_load_async_to_lds_b128 %0, %1, off"
                 :: "v"(lds), "v"(g) : "memory");
#else
    (void)g; (void)l;
#endif
}
__device__ __forceinline__ void async_wait0() {
#if defined(__HIP_DEVICE_COMPILE__)
    asm volatile("s_wait_asynccnt 0" ::: "memory");
#endif
}

__device__ __forceinline__ float wave_sum32(float s) {
    #pragma unroll
    for (int m = 1; m < 32; m <<= 1) s += __shfl_xor(s, m, 32);
    return s;
}
__device__ __forceinline__ float sgate(float v) {
    float sg = 1.f / (1.f + __expf(-10.f * (fabsf(v) - 0.3f)));
    return v * sg;
}

// ---------------- layout / elementwise kernels ----------------

__global__ void k_nchw2nhwc(const float* __restrict__ src, float* __restrict__ dst) {
    long i = (long)blockIdx.x * 256 + threadIdx.x;          // flat NHWC index
    int c = i & (C_ - 1);
    long bn = i >> 7;
    int n = (int)(bn % N_);
    int b = (int)(bn / N_);
    dst[i] = src[((long)b * C_ + c) * N_ + n];
}

__global__ void k_cvt_bf16(const float* __restrict__ s, __bf16* __restrict__ d, int n) {
    int i = blockIdx.x * 256 + threadIdx.x;
    if (i < n) d[i] = (__bf16)s[i];
}

// w [O,I,3,3] -> wpk [t][O][I], t = dy*3+dx
__global__ void k_pack_convw(const float* __restrict__ w, __bf16* __restrict__ wpk,
                             int O, int I) {
    int idx = blockIdx.x * 256 + threadIdx.x;
    if (idx >= O * I * 9) return;
    int t = idx / (O * I);
    int r = idx - t * (O * I);
    int o = r / I, c = r - o * I;
    int dy = t / 3, dx = t - dy * 3;
    wpk[idx] = (__bf16)w[((long)(o * I + c) * 3 + dy) * 3 + dx];
}

// out[b,n,c] = in[b,n,c] * sw[b,c]
__global__ void k_rowscale(const float* __restrict__ in, const float* __restrict__ sw,
                           float* __restrict__ out) {
    long i = (long)blockIdx.x * 256 + threadIdx.x;
    int c = i & (C_ - 1);
    int b = (int)((i >> 7) / N_);
    out[i] = in[i] * sw[b * C_ + c];
}

// out = base + sparse_gate(val)
__global__ void k_sgate_add(const float* __restrict__ base, const float* __restrict__ val,
                            float* __restrict__ out) {
    long i = (long)blockIdx.x * 256 + threadIdx.x;
    out[i] = base[i] + sgate(val[i]);
}

// d_out (NCHW) = x2 + sgate(alpha * ffn), reading NHWC buffers
__global__ void k_final(const float* __restrict__ x2, const float* __restrict__ ffn,
                        const float* __restrict__ alpha, float* __restrict__ out) {
    long i = (long)blockIdx.x * 256 + threadIdx.x;          // flat NCHW index
    int n = (int)(i % N_);
    long bc = i / N_;
    int c = (int)(bc % C_);
    int b = (int)(bc / C_);
    long j = ((long)b * N_ + n) * C_ + c;
    float v = ffn[j] * alpha[0];
    out[i] = x2[j] + sgate(v);
}

// ---------------- adaptive sparse norm ----------------

__global__ void k_asn_ln(const float* __restrict__ xin, const float* __restrict__ g,
                         const float* __restrict__ bt, float* __restrict__ xn) {
    int tid = threadIdx.x, wave = tid >> 5, lane = tid & 31;
    long pos = (long)blockIdx.x * 8 + wave;                 // flat b*N + n
    const float* xr = xin + pos * C_;
    float4 v = *(const float4*)(xr + lane * 4);
    float s  = v.x + v.y + v.z + v.w;
    float ss = v.x * v.x + v.y * v.y + v.z * v.z + v.w * v.w;
    s  = wave_sum32(s);
    ss = wave_sum32(ss);
    float mu = s * (1.f / C_);
    float var = ss * (1.f / C_) - mu * mu;
    float rstd = rsqrtf(var + 1e-5f);
    float4 gg = *(const float4*)(g + lane * 4);
    float4 bb = *(const float4*)(bt + lane * 4);
    float4 o;
    o.x = (v.x - mu) * rstd * gg.x + bb.x;
    o.y = (v.y - mu) * rstd * gg.y + bb.y;
    o.z = (v.z - mu) * rstd * gg.z + bb.z;
    o.w = (v.w - mu) * rstd * gg.w + bb.w;
    *(float4*)(xn + pos * C_ + lane * 4) = o;
}

__global__ void k_colsum(const float* __restrict__ xn, float* __restrict__ gsum) {
    int tid = threadIdx.x, wave = tid >> 5, lane = tid & 31;
    int o = blockIdx.x * 8 + wave;                          // b*C + c
    int b = o >> 7, c = o & (C_ - 1);
    const float* p = xn + (long)b * N_ * C_ + c;
    float s = 0.f;
    for (int n = lane; n < N_; n += 32) s += p[(long)n * C_];
    s = wave_sum32(s);
    if (lane == 0) gsum[o] = s;
}

__global__ void k_asn_gate(const float* __restrict__ gsum, const float* __restrict__ spw,
                           const float* __restrict__ spb, float* __restrict__ sw,
                           float* __restrict__ pooled) {
    int tid = threadIdx.x, wave = tid >> 5, lane = tid & 31;
    int o = blockIdx.x * 8 + wave;                          // b*C + c
    int b = o >> 7, c = o & (C_ - 1);
    float4 a4 = *(const float4*)(gsum + b * C_ + lane * 4);
    float4 w4 = *(const float4*)(spw + (long)c * C_ + lane * 4);
    float acc = a4.x * w4.x + a4.y * w4.y + a4.z * w4.z + a4.w * w4.w;
    acc = wave_sum32(acc) * (1.f / N_);
    if (lane == 0) {
        float sg = 1.f / (1.f + __expf(-(acc + spb[c])));
        sw[o] = sg;
        if (pooled) pooled[o] = sg * gsum[o] * (1.f / N_);
    }
}

__global__ void k_ffn_gate(const float* __restrict__ pooled, const float* __restrict__ gw,
                           const float* __restrict__ gb, float* __restrict__ g) {
    int tid = threadIdx.x, wave = tid >> 5, lane = tid & 31;
    int o = blockIdx.x * 8 + wave;                          // b*HID + hd
    int b = o >> 8, hd = o & (HID_ - 1);
    float4 p4 = *(const float4*)(pooled + b * C_ + lane * 4);
    float4 w4 = *(const float4*)(gw + (long)hd * C_ + lane * 4);
    float acc = p4.x * w4.x + p4.y * w4.y + p4.z * w4.z + p4.w * w4.w;
    acc = wave_sum32(acc);
    if (lane == 0) g[o] = 1.f / (1.f + __expf(-(acc + gb[hd])));
}

// ---------------- WMMA bf16 GEMM:  C[m,n] = A[m,:] . W[n,:] (+bias) ----------------
// A fp32 row-major [M,K]; W bf16 row-major [N,K]. Block tile 32x128, 8 waves (2x4),
// each wave 16x32 (2 accumulators). B tile staged via async global->LDS (bf16, no cvt).

__global__ void k_gemm_bf16(const float* __restrict__ A, const __bf16* __restrict__ W,
                            const float* __restrict__ bias, float* __restrict__ Cf,
                            __bf16* __restrict__ Cb, int M, int Nn, int K) {
    __shared__ __align__(16) __bf16 At[32][40];
    __shared__ __align__(16) __bf16 Bt[128][40];
    int tid = threadIdx.x, wave = tid >> 5, lane = tid & 31;
    int lh = lane >> 4, ll = lane & 15;
    int waveM = wave >> 2, waveN = wave & 3;
    int m0 = blockIdx.y * 32, n0 = blockIdx.x * 128;
    f32x8 acc0 = {}, acc1 = {};
    for (int kc = 0; kc < K; kc += 32) {
        // B tile: 128 rows x 32 halfs, async DMA into WMMA-ready LDS layout
        #pragma unroll
        for (int j = 0; j < 2; ++j) {
            int idx = tid * 2 + j;
            int row = idx >> 2, seg = idx & 3;
            async_copy_b128(W + (long)(n0 + row) * K + kc + seg * 8, &Bt[row][seg * 8]);
        }
        // A tile: fp32 -> bf16 via VALU (overlaps with async copy)
        {
            int row = tid >> 3, cg = (tid & 7) * 4;
            float4 v = *(const float4*)(A + (long)(m0 + row) * K + kc + cg);
            At[row][cg + 0] = (__bf16)v.x; At[row][cg + 1] = (__bf16)v.y;
            At[row][cg + 2] = (__bf16)v.z; At[row][cg + 3] = (__bf16)v.w;
        }
        async_wait0();
        __syncthreads();
        FragBF a, b0, b1;
        a.u[0]  = *(const uint4*)&At[waveM * 16 + ll][16 * lh];
        a.u[1]  = *(const uint4*)&At[waveM * 16 + ll][16 * lh + 8];
        b0.u[0] = *(const uint4*)&Bt[waveN * 32 + ll][16 * lh];
        b0.u[1] = *(const uint4*)&Bt[waveN * 32 + ll][16 * lh + 8];
        b1.u[0] = *(const uint4*)&Bt[waveN * 32 + 16 + ll][16 * lh];
        b1.u[1] = *(const uint4*)&Bt[waveN * 32 + 16 + ll][16 * lh + 8];
        acc0 = __builtin_amdgcn_wmma_f32_16x16x32_bf16(false, a.v, false, b0.v,
                                                       (short)0, acc0, false, false);
        acc1 = __builtin_amdgcn_wmma_f32_16x16x32_bf16(false, a.v, false, b1.v,
                                                       (short)0, acc1, false, false);
        __syncthreads();
    }
    int nw = n0 + waveN * 32;
    float bv0 = bias ? bias[nw + ll] : 0.f;
    float bv1 = bias ? bias[nw + 16 + ll] : 0.f;
    #pragma unroll
    for (int i = 0; i < 8; ++i) {
        int m = m0 + waveM * 16 + i + 8 * lh;
        float v0 = acc0[i] + bv0, v1 = acc1[i] + bv1;
        if (Cb) {
            Cb[(long)m * Nn + nw + ll]      = (__bf16)v0;
            Cb[(long)m * Nn + nw + 16 + ll] = (__bf16)v1;
        } else {
            Cf[(long)m * Nn + nw + ll]      = v0;
            Cf[(long)m * Nn + nw + 16 + ll] = v1;
        }
    }
}

// ---------------- implicit-GEMM 3x3 conv (SAME), NHWC, WMMA bf16 ----------------
// in fp32 [B,N,Cin]; wpk bf16 [9][Cout][Cin]; out fp32 [B,N,Cout]
// block: 32 positions x 128 out-channels; 8 waves (2x4), 2 accumulators per wave.

__global__ void k_conv3x3(const float* __restrict__ in, const __bf16* __restrict__ wpk,
                          const float* __restrict__ bias, const float* __restrict__ gate,
                          float* __restrict__ out, int Cin, int Cout, int relu) {
    __shared__ __align__(16) __bf16 At[32][40];
    __shared__ __align__(16) __bf16 Bt[128][40];
    int tid = threadIdx.x, wave = tid >> 5, lane = tid & 31;
    int lh = lane >> 4, ll = lane & 15;
    int waveM = wave >> 2, waveN = wave & 3;
    int b = blockIdx.z;
    int pp = blockIdx.x * 32;                 // flat position tile within image
    int n0 = blockIdx.y * 128;
    f32x8 acc0 = {}, acc1 = {};
    for (int t = 0; t < 9; ++t) {
        int dy = t / 3 - 1, dx = t % 3 - 1;
        for (int kc = 0; kc < Cin; kc += 32) {
            // B tile: pre-packed bf16 weights, async DMA
            #pragma unroll
            for (int j = 0; j < 2; ++j) {
                int idx = tid * 2 + j;
                int row = idx >> 2, seg = idx & 3;
                async_copy_b128(wpk + ((long)t * Cout + n0 + row) * Cin + kc + seg * 8,
                                &Bt[row][seg * 8]);
            }
            // A tile: zero-padded im2col gather, fp32 -> bf16
            {
                int p = tid >> 3, cg = (tid & 7) * 4;
                int pf = pp + p;
                int h2 = pf / W_ + dy, w2 = pf % W_ + dx;
                float4 v = make_float4(0.f, 0.f, 0.f, 0.f);
                if (h2 >= 0 && h2 < H_ && w2 >= 0 && w2 < W_) {
                    long q = (long)b * N_ + h2 * W_ + w2;
                    v = *(const float4*)(in + q * Cin + kc + cg);
                }
                At[p][cg + 0] = (__bf16)v.x; At[p][cg + 1] = (__bf16)v.y;
                At[p][cg + 2] = (__bf16)v.z; At[p][cg + 3] = (__bf16)v.w;
            }
            async_wait0();
            __syncthreads();
            FragBF a, b0, b1;
            a.u[0]  = *(const uint4*)&At[waveM * 16 + ll][16 * lh];
            a.u[1]  = *(const uint4*)&At[waveM * 16 + ll][16 * lh + 8];
            b0.u[0] = *(const uint4*)&Bt[waveN * 32 + ll][16 * lh];
            b0.u[1] = *(const uint4*)&Bt[waveN * 32 + ll][16 * lh + 8];
            b1.u[0] = *(const uint4*)&Bt[waveN * 32 + 16 + ll][16 * lh];
            b1.u[1] = *(const uint4*)&Bt[waveN * 32 + 16 + ll][16 * lh + 8];
            acc0 = __builtin_amdgcn_wmma_f32_16x16x32_bf16(false, a.v, false, b0.v,
                                                           (short)0, acc0, false, false);
            acc1 = __builtin_amdgcn_wmma_f32_16x16x32_bf16(false, a.v, false, b1.v,
                                                           (short)0, acc1, false, false);
            __syncthreads();
        }
    }
    int nw = n0 + waveN * 32;
    float bv0 = bias ? bias[nw + ll] : 0.f;
    float bv1 = bias ? bias[nw + 16 + ll] : 0.f;
    float gv0 = gate ? gate[b * Cout + nw + ll] : 1.f;
    float gv1 = gate ? gate[b * Cout + nw + 16 + ll] : 1.f;
    #pragma unroll
    for (int i = 0; i < 8; ++i) {
        int p = waveM * 16 + i + 8 * lh;
        float v0 = acc0[i] + bv0, v1 = acc1[i] + bv1;
        if (relu) { v0 = fmaxf(v0, 0.f); v1 = fmaxf(v1, 0.f); }
        v0 *= gv0; v1 *= gv1;
        long pos = (long)b * N_ + pp + p;
        out[pos * Cout + nw + ll]      = v0;
        out[pos * Cout + nw + 16 + ll] = v1;
    }
}

// ---------------- flash attention with WMMA (bf16 QKV) ----------------
// qkv bf16 [B,N,3C]; attn fp32 [B,N,C]; local 3x3 window masked OUT.
// grid (N/128, HEADS, B); 8 waves, one 16-query tile each; 32-key panels.
// K panels async-copied straight into WMMA B-operand LDS layout.

__global__ void k_flash(const __bf16* __restrict__ qkv, float* __restrict__ attn) {
    __shared__ __align__(16) __bf16 Kt[32][40];         // [key][d]
    __shared__ __align__(16) __bf16 Vt[32][40];         // [d][key] (transposed)
    __shared__ __align__(16) __bf16 Pt[8][16][40];      // per-wave P relayout
    int tid = threadIdx.x, wave = tid >> 5, lane = tid & 31;
    int lh = lane >> 4, ll = lane & 15;
    int b = blockIdx.z, h = blockIdx.y;
    int q0 = blockIdx.x * 128 + wave * 16;
    const float sc = 0.17677669529663687f;              // 1/sqrt(32), folded into scores

    FragBF qf;
    {
        const __bf16* qr = qkv + ((long)b * N_ + q0 + ll) * (3 * C_) + h * DH_;
        qf.u[0] = *(const uint4*)(qr + 16 * lh);
        qf.u[1] = *(const uint4*)(qr + 16 * lh + 8);
    }
    f32x8 oa = {}, ob = {};
    float mrow[8], lrow[8];
    int qri[8], qrj[8];
    #pragma unroll
    for (int i = 0; i < 8; ++i) {
        mrow[i] = -1e30f; lrow[i] = 0.f;
        int q = q0 + i + 8 * lh;
        qri[i] = q / W_; qrj[i] = q % W_;
    }

    for (int j0 = 0; j0 < N_; j0 += 32) {
        __syncthreads();
        // K panel: async DMA, already in B-operand layout (32 keys x 32 d)
        if (tid < 128) {
            int key = tid >> 2, seg = tid & 3;
            async_copy_b128(qkv + ((long)b * N_ + j0 + key) * (3 * C_) + C_ + h * DH_ + seg * 8,
                            &Kt[key][seg * 8]);
        }
        // V panel: bf16 load + transposed LDS store
        {
            int key = tid >> 3, dg = (tid & 7) * 4;
            union { uint2 u; __bf16 e[4]; } vv;
            vv.u = *(const uint2*)(qkv + ((long)b * N_ + j0 + key) * (3 * C_) + 2 * C_ + h * DH_ + dg);
            Vt[dg + 0][key] = vv.e[0]; Vt[dg + 1][key] = vv.e[1];
            Vt[dg + 2][key] = vv.e[2]; Vt[dg + 3][key] = vv.e[3];
        }
        async_wait0();
        __syncthreads();
        f32x8 z = {};
        FragBF kb;
        kb.u[0] = *(const uint4*)&Kt[ll][16 * lh];
        kb.u[1] = *(const uint4*)&Kt[ll][16 * lh + 8];
        f32x8 sa = __builtin_amdgcn_wmma_f32_16x16x32_bf16(false, qf.v, false, kb.v,
                                                           (short)0, z, false, false);
        kb.u[0] = *(const uint4*)&Kt[16 + ll][16 * lh];
        kb.u[1] = *(const uint4*)&Kt[16 + ll][16 * lh + 8];
        f32x8 sb = __builtin_amdgcn_wmma_f32_16x16x32_bf16(false, qf.v, false, kb.v,
                                                           (short)0, z, false, false);
        int ka = j0 + ll, kb2 = ka + 16;
        int kai = ka / W_, kaj = ka % W_, kbi = kb2 / W_, kbj = kb2 % W_;
        #pragma unroll
        for (int i = 0; i < 8; ++i) {
            float va = sa[i] * sc, vb = sb[i] * sc;
            int d0 = qri[i] - kai; if (d0 < 0) d0 = -d0;
            int d1 = qrj[i] - kaj; if (d1 < 0) d1 = -d1;
            int d2 = qri[i] - kbi; if (d2 < 0) d2 = -d2;
            int d3 = qrj[i] - kbj; if (d3 < 0) d3 = -d3;
            if (d0 <= 1 && d1 <= 1) va = -1e30f;
            if (d2 <= 1 && d3 <= 1) vb = -1e30f;
            float mx = fmaxf(va, vb);
            #pragma unroll
            for (int s = 1; s < 16; s <<= 1) mx = fmaxf(mx, __shfl_xor(mx, s, 32));
            float mn = fmaxf(mrow[i], mx);
            float corr = __expf(mrow[i] - mn);
            float pa = __expf(va - mn), pb = __expf(vb - mn);
            float rs = pa + pb;
            #pragma unroll
            for (int s = 1; s < 16; s <<= 1) rs += __shfl_xor(rs, s, 32);
            lrow[i] = lrow[i] * corr + rs;
            mrow[i] = mn;
            oa[i] *= corr; ob[i] *= corr;
            int r = i + 8 * lh;
            Pt[wave][r][ll]      = (__bf16)pa;
            Pt[wave][r][16 + ll] = (__bf16)pb;
        }
        __syncthreads();
        FragBF pf, vf;
        pf.u[0] = *(const uint4*)&Pt[wave][ll][16 * lh];
        pf.u[1] = *(const uint4*)&Pt[wave][ll][16 * lh + 8];
        vf.u[0] = *(const uint4*)&Vt[ll][16 * lh];
        vf.u[1] = *(const uint4*)&Vt[ll][16 * lh + 8];
        oa = __builtin_amdgcn_wmma_f32_16x16x32_bf16(false, pf.v, false, vf.v,
                                                     (short)0, oa, false, false);
        vf.u[0] = *(const uint4*)&Vt[16 + ll][16 * lh];
        vf.u[1] = *(const uint4*)&Vt[16 + ll][16 * lh + 8];
        ob = __builtin_amdgcn_wmma_f32_16x16x32_bf16(false, pf.v, false, vf.v,
                                                     (short)0, ob, false, false);
    }
    #pragma unroll
    for (int i = 0; i < 8; ++i) {
        int qrow = q0 + i + 8 * lh;
        float inv = 1.f / lrow[i];
        float* dst = attn + ((long)b * N_ + qrow) * C_ + h * DH_;
        dst[ll]      = oa[i] * inv;
        dst[16 + ll] = ob[i] * inv;
    }
}

// ---------------- host orchestration ----------------

extern "C" void kernel_launch(void* const* d_in, const int* in_sizes, int n_in,
                              void* d_out, int out_size, void* d_ws, size_t ws_size,
                              hipStream_t stream) {
    (void)in_sizes; (void)n_in; (void)out_size; (void)ws_size;
    const float* x     = (const float*)d_in[0];
    const float* ln1g  = (const float*)d_in[1];
    const float* ln1b  = (const float*)d_in[2];
    const float* sp1w  = (const float*)d_in[3];
    const float* sp1b  = (const float*)d_in[4];
    const float* ln2g  = (const float*)d_in[5];
    const float* ln2b  = (const float*)d_in[6];
    const float* sp2w  = (const float*)d_in[7];
    const float* sp2b  = (const float*)d_in[8];
    const float* inpw  = (const float*)d_in[9];
    const float* inpb  = (const float*)d_in[10];
    const float* outw  = (const float*)d_in[11];
    const float* outb  = (const float*)d_in[12];
    const float* loww  = (const float*)d_in[13];
    const float* c1w   = (const float*)d_in[14];
    const float* c1b   = (const float*)d_in[15];
    const float* c2w   = (const float*)d_in[16];
    const float* c2b   = (const float*)d_in[17];
    const float* gw    = (const float*)d_in[18];
    const float* gb    = (const float*)d_in[19];
    const float* cv1w  = (const float*)d_in[20];
    const float* cv2w  = (const float*)d_in[21];
    const float* alpha = (const float*)d_in[22];

    char* base = (char*)d_ws;
    size_t off = 0;
    auto alloc = [&](size_t bytes) -> char* {
        char* p = base + off;
        off += (bytes + 255) & ~(size_t)255;
        return p;
    };
    float*  xT    = (float*)alloc((size_t)BNC_ * 4);
    float*  xn    = (float*)alloc((size_t)BNC_ * 4);
    float*  gsum  = (float*)alloc((size_t)B_ * C_ * 4);
    float*  sw1   = (float*)alloc((size_t)B_ * C_ * 4);
    float*  sw2   = (float*)alloc((size_t)B_ * C_ * 4);
    float*  pooled= (float*)alloc((size_t)B_ * C_ * 4);
    float*  xns   = (float*)alloc((size_t)BNC_ * 4);
    __bf16* qkvb  = (__bf16*)alloc((size_t)BN_ * 3 * C_ * 2);
    float*  attnb = (float*)alloc((size_t)BNC_ * 4);
    float*  proj  = (float*)alloc((size_t)BNC_ * 4);
    float*  x1    = (float*)alloc((size_t)BNC_ * 4);
    float*  lf    = (float*)alloc((size_t)BNC_ * 4);
    float*  tmpc  = (float*)alloc((size_t)BNC_ * 4);
    float*  comp  = (float*)alloc((size_t)BNC_ * 4);
    float*  x2v   = (float*)alloc((size_t)BNC_ * 4);
    float*  gvec  = (float*)alloc((size_t)B_ * HID_ * 4);
    float*  hbuf  = (float*)alloc((size_t)BN_ * HID_ * 4);
    float*  ffn   = (float*)alloc((size_t)BNC_ * 4);
    __bf16* wq    = (__bf16*)alloc((size_t)3 * C_ * C_ * 2);
    __bf16* wo    = (__bf16*)alloc((size_t)C_ * C_ * 2);
    __bf16* wpL   = (__bf16*)alloc((size_t)9 * C_ * C_ * 2);
    __bf16* wp1   = (__bf16*)alloc((size_t)9 * C_ * C_ * 2);
    __bf16* wp2   = (__bf16*)alloc((size_t)9 * C_ * C_ * 2);
    __bf16* wpc1  = (__bf16*)alloc((size_t)9 * HID_ * C_ * 2);
    __bf16* wpc2  = (__bf16*)alloc((size_t)9 * C_ * HID_ * 2);

    const dim3 blk(256);
    const int ewGrid = BNC_ / 256;             // 4608

    // weight prep (bf16 conversions / conv packs)
    k_cvt_bf16<<<(3 * C_ * C_) / 256, blk, 0, stream>>>(inpw, wq, 3 * C_ * C_);
    k_cvt_bf16<<<(C_ * C_) / 256, blk, 0, stream>>>(outw, wo, C_ * C_);
    k_pack_convw<<<(9 * C_ * C_) / 256, blk, 0, stream>>>(loww, wpL, C_, C_);
    k_pack_convw<<<(9 * C_ * C_) / 256, blk, 0, stream>>>(c1w, wp1, C_, C_);
    k_pack_convw<<<(9 * C_ * C_) / 256, blk, 0, stream>>>(c2w, wp2, C_, C_);
    k_pack_convw<<<(9 * HID_ * C_) / 256, blk, 0, stream>>>(cv1w, wpc1, HID_, C_);
    k_pack_convw<<<(9 * C_ * HID_) / 256, blk, 0, stream>>>(cv2w, wpc2, C_, HID_);

    // NCHW -> NHWC
    k_nchw2nhwc<<<ewGrid, blk, 0, stream>>>(x, xT);

    // --- adaptive sparse norm 1 + attention + residual ---
    k_asn_ln<<<BN_ / 8, blk, 0, stream>>>(xT, ln1g, ln1b, xn);
    k_colsum<<<(B_ * C_) / 8, blk, 0, stream>>>(xn, gsum);
    k_asn_gate<<<(B_ * C_) / 8, blk, 0, stream>>>(gsum, sp1w, sp1b, sw1, nullptr);
    k_rowscale<<<ewGrid, blk, 0, stream>>>(xn, sw1, xns);
    k_gemm_bf16<<<dim3((3 * C_) / 128, BN_ / 32), blk, 0, stream>>>(
        xns, wq, inpb, nullptr, qkvb, BN_, 3 * C_, C_);
    k_flash<<<dim3(N_ / 128, NH_, B_), blk, 0, stream>>>(qkvb, attnb);
    k_gemm_bf16<<<dim3(C_ / 128, BN_ / 32), blk, 0, stream>>>(
        attnb, wo, outb, proj, nullptr, BN_, C_, C_);
    k_sgate_add<<<ewGrid, blk, 0, stream>>>(xT, proj, x1);

    // --- low-frequency conv path ---
    k_conv3x3<<<dim3(N_ / 32, 1, B_), blk, 0, stream>>>(x1, wpL, nullptr, nullptr, lf, C_, C_, 0);
    k_conv3x3<<<dim3(N_ / 32, 1, B_), blk, 0, stream>>>(lf, wp1, c1b, nullptr, tmpc, C_, C_, 1);
    k_conv3x3<<<dim3(N_ / 32, 1, B_), blk, 0, stream>>>(tmpc, wp2, c2b, nullptr, comp, C_, C_, 0);
    k_sgate_add<<<ewGrid, blk, 0, stream>>>(x1, comp, x2v);

    // --- adaptive sparse norm 2 + gated FFN ---
    k_asn_ln<<<BN_ / 8, blk, 0, stream>>>(x2v, ln2g, ln2b, xn);
    k_colsum<<<(B_ * C_) / 8, blk, 0, stream>>>(xn, gsum);
    k_asn_gate<<<(B_ * C_) / 8, blk, 0, stream>>>(gsum, sp2w, sp2b, sw2, pooled);
    k_rowscale<<<ewGrid, blk, 0, stream>>>(xn, sw2, xns);
    k_ffn_gate<<<(B_ * HID_) / 8, blk, 0, stream>>>(pooled, gw, gb, gvec);
    k_conv3x3<<<dim3(N_ / 32, HID_ / 128, B_), blk, 0, stream>>>(
        xns, wpc1, nullptr, gvec, hbuf, C_, HID_, 0);
    k_conv3x3<<<dim3(N_ / 32, 1, B_), blk, 0, stream>>>(
        hbuf, wpc2, nullptr, nullptr, ffn, HID_, C_, 0);

    // --- final residual + NHWC -> NCHW into d_out ---
    k_final<<<ewGrid, blk, 0, stream>>>(x2v, ffn, alpha, (float*)d_out);
}
